// GCN_24257975288005
// MI455X (gfx1250) — matrix operand order repaired and checked by reading, not compiled, not run on hardware
//
#include <hip/hip_runtime.h>
#include <hip/hip_bf16.h>

#define NNODES 100000
#define NEDGES 1600000
#define EPSV 1e-5f

typedef __bf16 bf16_t;
typedef __attribute__((ext_vector_type(16))) __bf16 v16bf;
typedef __attribute__((ext_vector_type(8)))  float  v8f;

union Afrag { v16bf v; unsigned u[8]; };
union Bfrag { v16bf v; uint4 q[2]; };

// ---------------------------------------------------------------------------
// fp32 -> bf16 (round-to-nearest-even), packed pair -> one dword
// ---------------------------------------------------------------------------
__device__ __forceinline__ unsigned bf16bits(float f) {
  union { float f; unsigned u; } in; in.f = f;
  unsigned u = in.u;
  u += 0x7FFFu + ((u >> 16) & 1u);
  return u >> 16;
}
__device__ __forceinline__ unsigned pack_bf16x2(float lo, float hi) {
  return bf16bits(lo) | (bf16bits(hi) << 16);
}

__device__ __forceinline__ float wave_reduce_add(float v) {
  #pragma unroll
  for (int m = 16; m > 0; m >>= 1) v += __shfl_xor(v, m, 32);
  return v;
}

template <int NKB>
__device__ __forceinline__ void loadB(Bfrag (&bb)[NKB], const uint4* base) {
  #pragma unroll
  for (int kb = 0; kb < NKB; ++kb) {
    bb[kb].q[0] = base[kb * 64];
    bb[kb].q[1] = base[kb * 64 + 1];
  }
}

template <int NKB>
__device__ __forceinline__ v8f wmma_chain(const Afrag (&a)[NKB], const Bfrag (&bb)[NKB]) {
  v8f acc = {};
  #pragma unroll
  for (int kb = 0; kb < NKB; ++kb) {
    acc = __builtin_amdgcn_wmma_f32_16x16x32_bf16(
        false, a[kb].v, false, bb[kb].v, (short)0, acc, false, false);
  }
  return acc;
}

// ---------------------------------------------------------------------------
// Pre-pack W[K, nout] (fp32 row-major) into per-lane B-fragment layout:
//   frag f = ct*nkb + kb ; dword index = (f*32 + lane)*8 + j
//   lane: col = ct*16 + (lane&15), kc = kb*32 + (lane>>4)*16
//   dword j holds bf16 pair (K = kc+2j, kc+2j+1); zero padded OOB.
// ---------------------------------------------------------------------------
__global__ void pack_w_kernel(const float* __restrict__ W, unsigned* __restrict__ Wp,
                              int K, int nout, int nct, int nkb)
{
  int idx = blockIdx.x * blockDim.x + threadIdx.x;
  int total = nct * nkb * 32 * 8;
  if (idx >= total) return;
  int j    = idx & 7;
  int lane = (idx >> 3) & 31;
  int f    = idx >> 8;
  int kb   = f % nkb;
  int ct   = f / nkb;
  int col  = ct * 16 + (lane & 15);
  int kc   = kb * 32 + (lane >> 4) * 16 + 2 * j;
  float w0 = (col < nout && kc     < K) ? W[(long long)kc * nout + col]       : 0.0f;
  float w1 = (col < nout && kc + 1 < K) ? W[(long long)(kc + 1) * nout + col] : 0.0f;
  Wp[idx] = pack_bf16x2(w0, w1);
}

// ---------------------------------------------------------------------------
// GEMM: H[nrows, nout] = X[nrows, K] @ W[K, nout] via v_wmma_f32_16x16x32_bf16
// 256 threads = 8 waves; 128-row block tile. A staged in LDS as bf16 pairs
// and held in registers across the column loop. Column tiles are software
// pipelined (ping-pong b0/b1): B fragments for tile ct+1 are loaded while the
// WMMA chain consumes tile ct's fragments loaded last iteration, so the
// matrix ops never wait on in-flight loads.
// ---------------------------------------------------------------------------
template <int KPAD>
__global__ __launch_bounds__(256) void gemm_wmma_t(
    const float* __restrict__ X, const unsigned* __restrict__ Wp,
    float* __restrict__ H, int nrows, int K, int nout)
{
  constexpr int NKB = KPAD / 32;
  constexpr int KP2 = KPAD / 2;
  __shared__ unsigned As[128 * KP2];     // 128 rows x KPAD bf16 as packed pairs
  const int tid = threadIdx.x;
  const int rowBase = blockIdx.x * 128;

  // ---- stage X tile into LDS as packed bf16 pairs (K even; zero padded) ----
  for (int i = tid; i < 128 * KP2; i += 256) {
    int r  = i / KP2;                    // constant divisor
    int kp = i - r * KP2;
    int k  = kp * 2;
    int row = rowBase + r;
    float2 v = make_float2(0.0f, 0.0f);
    if (row < nrows && k < K) v = *(const float2*)(X + (long long)row * K + k);
    As[i] = pack_bf16x2(v.x, v.y);
  }
  __syncthreads();

  const int wave = tid >> 5;             // owns rows [wave*16, wave*16+16)
  const int lane = tid & 31;
  const int lm   = lane & 15;
  const int hi   = lane >> 4;
  const unsigned* Arow = &As[(wave * 16 + lm) * KP2];

  // A fragments: invariant across ct -> kept resident in VGPRs
  Afrag a[NKB];
  #pragma unroll
  for (int kb = 0; kb < NKB; ++kb) {
    const int pbase = kb * 16 + hi * 8;  // packed-pair base index
    #pragma unroll
    for (int j = 0; j < 4; ++j) {
      a[kb].u[j]     = Arow[pbase + j];
      a[kb].u[j + 4] = Arow[pbase + 4 + j];
    }
  }

  const bool fullTile = (rowBase + 128 <= nrows);
  const int nct = (nout + 15) >> 4;
  const uint4* wbase = (const uint4*)Wp + lane * 2;
  auto bptr = [&](int ct) { return wbase + (long long)ct * (NKB * 64); };

  // C/D layout: lane -> column (lm), VGPR j -> row j + 8*hi
  auto store_tile = [&](int ct, const v8f& acc) {
    const int col = ct * 16 + lm;
    if (col < nout) {
      float* Hc = H + (long long)(rowBase + wave * 16 + hi * 8) * nout + col;
      if (fullTile) {
        #pragma unroll
        for (int j = 0; j < 8; ++j) Hc[(long long)j * nout] = acc[j];
      } else {
        #pragma unroll
        for (int j = 0; j < 8; ++j) {
          if (rowBase + wave * 16 + hi * 8 + j < nrows) Hc[(long long)j * nout] = acc[j];
        }
      }
    }
  };

  Bfrag b0[NKB], b1[NKB];
  loadB<NKB>(b0, bptr(0));                       // prologue
  for (int ct = 0; ct < nct; ct += 2) {
    const int ct1 = (ct + 1 < nct) ? ct + 1 : ct;      // clamped (redundant ok)
    loadB<NKB>(b1, bptr(ct1));                         // in-flight during chain
    __builtin_prefetch(bptr(ct + 2), 0, 3);            // near-cache prefetch
    v8f acc = wmma_chain<NKB>(a, b0);
    store_tile(ct, acc);
    if (ct + 1 < nct) {
      const int ct2 = (ct + 2 < nct) ? ct + 2 : ct1;   // clamped
      loadB<NKB>(b0, bptr(ct2));                       // in-flight during chain
      acc = wmma_chain<NKB>(a, b1);
      store_tile(ct1, acc);
    }
  }
}

// ---------------------------------------------------------------------------
// Normalization coefficients
// ---------------------------------------------------------------------------
__global__ void deg_init_kernel(float* deg, int n) {
  int i = blockIdx.x * blockDim.x + threadIdx.x;
  if (i < n) deg[i] = 1.0f;                       // self-loop contribution
}

__global__ void deg_count_kernel(const int* __restrict__ dst, float* deg, int nE) {
  int e = blockIdx.x * blockDim.x + threadIdx.x;
  if (e < nE) atomicAdd(&deg[dst[e]], 1.0f);
}

__global__ void dinv_kernel(float* deg, float* dinv2, int n) {
  int i = blockIdx.x * blockDim.x + threadIdx.x;
  if (i < n) {
    float r = rsqrtf(fmaxf(deg[i], 1.0f));        // deg >= 1 always
    deg[i] = r;                                   // deg[] becomes dinv[]
    dinv2[i] = r * r;                             // self-loop norm
  }
}

__global__ void norm_kernel(const int* __restrict__ src, const int* __restrict__ dst,
                            const float* __restrict__ dinv, float* __restrict__ nrm, int nE) {
  int e = blockIdx.x * blockDim.x + threadIdx.x;
  if (e < nE) nrm[e] = dinv[src[e]] * dinv[dst[e]];
}

// ---------------------------------------------------------------------------
// Message passing: agg[dst] += norm_e * H[src]  (32 lanes/edge, 8 edges/block)
// ---------------------------------------------------------------------------
__global__ __launch_bounds__(256) void edge_scatter_kernel(
    const float* __restrict__ H, const int* __restrict__ src,
    const int* __restrict__ dst, const float* __restrict__ nrm,
    float* __restrict__ out, int nE, int d)
{
  int e = blockIdx.x * 8 + (threadIdx.x >> 5);
  int lane = threadIdx.x & 31;
  if (e >= nE) return;
  int s = src[e], t = dst[e];
  float w = nrm[e];
  const float* hs = H + (long long)s * d;
  float* od = out + (long long)t * d;
  for (int f = lane; f < d; f += 32)
    atomicAdd(&od[f], hs[f] * w);
}

// self-loop: agg[i] += dinv2[i] * H[i]  (wave per node, no divisions)
__global__ __launch_bounds__(256) void self_add_kernel(
    const float* __restrict__ H, const float* __restrict__ dinv2,
    float* __restrict__ out, int n, int d)
{
  int nd = blockIdx.x * 8 + (threadIdx.x >> 5);
  int lane = threadIdx.x & 31;
  if (nd >= n) return;
  float w = dinv2[nd];
  const float* hr = H + (long long)nd * d;
  float* od = out + (long long)nd * d;
  for (int f = lane; f < d; f += 32)
    od[f] += hr[f] * w;
}

// ---------------------------------------------------------------------------
// Fused bias + ReLU + column sum/sumsq. Lanes of a wave hold the same column
// of 32 consecutive rows -> cross-lane shfl_xor reduce, then 2 LDS atomics
// per wave per column, then two global atomics per column per block.
// ---------------------------------------------------------------------------
__global__ __launch_bounds__(256) void bias_relu_stats_kernel(
    const float* __restrict__ agg, const float* __restrict__ bias,
    float* __restrict__ y, float* __restrict__ stats, int n, int d)
{
  extern __shared__ float sm[];                  // 2*d floats
  for (int i = threadIdx.x; i < 2 * d; i += 256) sm[i] = 0.0f;
  __syncthreads();
  int row = blockIdx.x * 256 + threadIdx.x;
  int lane = threadIdx.x & 31;
  const bool live = (row < n);
  const float* ar = agg + (long long)row * d;
  float* yr = y + (long long)row * d;
  for (int c = 0; c < d; ++c) {
    float v = 0.0f;
    if (live) {
      v = fmaxf(ar[c] + bias[c], 0.0f);
      yr[c] = v;
    }
    float s1 = wave_reduce_add(v);
    float s2 = wave_reduce_add(v * v);
    if (lane == 0) {
      atomicAdd(&sm[c], s1);
      atomicAdd(&sm[d + c], s2);
    }
  }
  __syncthreads();
  for (int i = threadIdx.x; i < 2 * d; i += 256) atomicAdd(&stats[i], sm[i]);
}

// wave per row; stats broadcast through cache
__global__ __launch_bounds__(256) void bn_apply_kernel(
    const float* __restrict__ y, const float* __restrict__ stats,
    const float* __restrict__ g, const float* __restrict__ beta,
    float* __restrict__ xo, int n, int d)
{
  int row = blockIdx.x * 8 + (threadIdx.x >> 5);
  int lane = threadIdx.x & 31;
  if (row >= n) return;
  const float invn = 1.0f / (float)n;
  const float* yr = y + (long long)row * d;
  float* xr = xo + (long long)row * d;
  for (int c = lane; c < d; c += 32) {
    float m = stats[c] * invn;
    float var = stats[d + c] * invn - m * m;
    xr[c] = (yr[c] - m) * rsqrtf(var + EPSV) * g[c] + beta[c];
  }
}

// ---------------------------------------------------------------------------
// Final layer: bias + log_softmax over d (= 17) columns, one thread per row
// ---------------------------------------------------------------------------
__global__ void bias_logsoftmax_kernel(const float* __restrict__ agg,
                                       const float* __restrict__ bias,
                                       float* __restrict__ out, int n, int d)
{
  int row = blockIdx.x * blockDim.x + threadIdx.x;
  if (row >= n) return;
  float vals[32];
  float mx = -3.4e38f;
  const float* ar = agg + (long long)row * d;
  for (int c = 0; c < d; ++c) {
    float v = ar[c] + bias[c];
    vals[c] = v;
    mx = fmaxf(mx, v);
  }
  float s = 0.0f;
  for (int c = 0; c < d; ++c) s += expf(vals[c] - mx);
  float ls = logf(s);
  float* orow = out + (long long)row * d;
  for (int c = 0; c < d; ++c) orow[c] = vals[c] - mx - ls;
}

__global__ void fill_zero_kernel(float* p, long long count) {
  long long i = (long long)blockIdx.x * blockDim.x + threadIdx.x;
  long long stride = (long long)gridDim.x * blockDim.x;
  for (; i < count; i += stride) p[i] = 0.0f;
}

// ---------------------------------------------------------------------------
// Orchestration
// ---------------------------------------------------------------------------
static void launch_gemm(int Kpad, const float* X, const unsigned* Wp, float* H,
                        int nrows, int K, int nout, hipStream_t stream)
{
  dim3 g((nrows + 127) / 128), b(256);
  switch (Kpad) {
    case 224: gemm_wmma_t<224><<<g, b, 0, stream>>>(X, Wp, H, nrows, K, nout); break;
    case 192: gemm_wmma_t<192><<<g, b, 0, stream>>>(X, Wp, H, nrows, K, nout); break;
    case 160: gemm_wmma_t<160><<<g, b, 0, stream>>>(X, Wp, H, nrows, K, nout); break;
    case 128: gemm_wmma_t<128><<<g, b, 0, stream>>>(X, Wp, H, nrows, K, nout); break;
    case  96: gemm_wmma_t< 96><<<g, b, 0, stream>>>(X, Wp, H, nrows, K, nout); break;
    case  64: gemm_wmma_t< 64><<<g, b, 0, stream>>>(X, Wp, H, nrows, K, nout); break;
    default:  gemm_wmma_t< 32><<<g, b, 0, stream>>>(X, Wp, H, nrows, K, nout); break;
  }
}

extern "C" void kernel_launch(void* const* d_in, const int* in_sizes, int n_in,
                              void* d_out, int out_size, void* d_ws, size_t ws_size,
                              hipStream_t stream)
{
  (void)in_sizes; (void)n_in; (void)out_size; (void)ws_size;
  const int N = NNODES, E = NEDGES;
  static const int D[7] = {220, 220, 150, 100, 60, 30, 17};

  const float* x  = (const float*)d_in[0];
  const int*   ei = (const int*)d_in[1];
  const int*  src = ei;              // edge_index[0]
  const int*  dst = ei + E;          // edge_index[1]
  const float *Wm[6], *bm[6], *gm[5], *betam[5];
  for (int i = 0; i < 6; ++i) {
    Wm[i] = (const float*)d_in[2 + 2 * i];
    bm[i] = (const float*)d_in[3 + 2 * i];
  }
  for (int i = 0; i < 5; ++i) {
    gm[i]    = (const float*)d_in[14 + 2 * i];
    betam[i] = (const float*)d_in[15 + 2 * i];
  }

  // workspace layout (floats / dwords)
  float* ws       = (float*)d_ws;
  float* deg      = ws;                              // N (becomes dinv)
  float* dinv2    = deg + N;                         // N
  float* nrm      = dinv2 + N;                       // E
  float* stats    = nrm + E;                         // 512
  unsigned* wpack = (unsigned*)(stats + 512);        // 32768 dwords, 16B aligned
  float* bufX     = (float*)(wpack + 32768);         // N*224
  float* bufH     = bufX + (long long)N * 224;       // N*224
  float* bufAgg   = bufH + (long long)N * 224;       // N*224

  // GCN normalization coefficients
  deg_init_kernel <<<(N + 255) / 256, 256, 0, stream>>>(deg, N);
  deg_count_kernel<<<(E + 255) / 256, 256, 0, stream>>>(dst, deg, E);
  dinv_kernel     <<<(N + 255) / 256, 256, 0, stream>>>(deg, dinv2, N);
  norm_kernel     <<<(E + 255) / 256, 256, 0, stream>>>(src, dst, deg, nrm, E);

  const float* Xcur = x;
  for (int L = 0; L < 6; ++L) {
    const int K = D[L], M = D[L + 1];
    const int Kpad = (K + 31) & ~31;
    const int nkb = Kpad / 32;
    const int nct = (M + 15) / 16;
    const long long tot = (long long)N * M;

    // pre-pack W into B-fragment order (bf16 pairs)
    const int ptotal = nct * nkb * 32 * 8;
    pack_w_kernel<<<(ptotal + 255) / 256, 256, 0, stream>>>(Wm[L], wpack, K, M, nct, nkb);

    launch_gemm(Kpad, Xcur, wpack, bufH, N, K, M, stream);

    fill_zero_kernel   <<<2048, 256, 0, stream>>>(bufAgg, tot);
    edge_scatter_kernel<<<(E + 7) / 8, 256, 0, stream>>>(bufH, src, dst, nrm, bufAgg, E, M);
    self_add_kernel    <<<(N + 7) / 8, 256, 0, stream>>>(bufH, dinv2, bufAgg, N, M);

    if (L < 5) {
      fill_zero_kernel<<<4, 256, 0, stream>>>(stats, 2 * M);
      bias_relu_stats_kernel<<<(N + 255) / 256, 256, (size_t)(2 * M) * sizeof(float), stream>>>(
          bufAgg, bm[L], bufH /* y reuses H */, stats, N, M);
      bn_apply_kernel<<<(N + 7) / 8, 256, 0, stream>>>(bufH, stats, gm[L], betam[L], bufX, N, M);
      Xcur = bufX;
    } else {
      bias_logsoftmax_kernel<<<(N + 255) / 256, 256, 0, stream>>>(bufAgg, bm[L], (float*)d_out, N, M);
    }
  }
}
